// AqSolModel_22333829939473
// MI455X (gfx1250) — compile-verified, matrix operand order checked by reading, not run on hardware
//
#include <hip/hip_runtime.h>
#include <hip/hip_bf16.h>

typedef __attribute__((ext_vector_type(16))) _Float16 v16h;
typedef __attribute__((ext_vector_type(8)))  _Float16 v8h;
typedef __attribute__((ext_vector_type(4)))  _Float16 v4h;
typedef __attribute__((ext_vector_type(8)))  float    v8f;

#define HID 256
#define LDA 40
#define LDB 40

// ---------------- utility kernels ----------------

__global__ void k_set_f32(float* p, float v, int n) {
  int i = blockIdx.x * blockDim.x + threadIdx.x;
  if (i < n) p[i] = v;
}

__global__ void k_deg_edges(const int* __restrict__ dst, float* __restrict__ deg, int E) {
  int e = blockIdx.x * blockDim.x + threadIdx.x;
  if (e < E) atomicAdd(&deg[dst[e]], 1.0f);
}

__global__ void k_rsqrt(const float* __restrict__ deg, float* __restrict__ dis, int n) {
  int i = blockIdx.x * blockDim.x + threadIdx.x;
  if (i < n) { float d = deg[i]; dis[i] = d > 0.f ? rsqrtf(d) : 0.f; }
}

__global__ void k_f32_to_f16(const float* __restrict__ in, _Float16* __restrict__ out, int n) {
  int i = blockIdx.x * blockDim.x + threadIdx.x;
  if (i < n) out[i] = (_Float16)in[i];
}

// ---------------- WMMA GEMM: C[M,F] = A[M,K](f16) * B[K,F](f16) (+bias,relu) ----------------
// Block: 256 threads = 8 waves, tile 32(M) x 64(F). Wave (mt,nt) does one 16x16 tile,
// one v_wmma_f32_16x16x32_f16 per 32-wide K step. Fragments follow ISA 7.12.2 layouts.

__global__ __launch_bounds__(256)
void k_gemm_wmma(const _Float16* __restrict__ A, const _Float16* __restrict__ B,
                 const float* __restrict__ bias, float* __restrict__ C,
                 int M, int K, int F, int relu) {
  __shared__ __align__(16) _Float16 sa[32 * LDA];
  __shared__ __align__(16) _Float16 sb[64 * LDB];

  const int t        = threadIdx.x;
  const int wid      = t >> 5;
  const int lane     = t & 31;
  const int half_sel = lane >> 4;   // 0: lanes 0-15, 1: lanes 16-31
  const int mn       = lane & 15;
  const int mt       = wid >> 2;    // 0..1  (M sub-tile)
  const int nt       = wid & 3;     // 0..3  (N sub-tile)
  const int m0       = blockIdx.x * 32;
  const int n0       = blockIdx.y * 64;

  // A-tile staging indices (32 rows x 32 k, v4h per thread)
  const int am = t >> 3;
  const int ak = (t & 7) * 4;
  int arow = m0 + am; if (arow > M - 1) arow = M - 1;   // clamp (stores are predicated)
  // B-tile staging indices (32 k x 64 n, transposed into LDS as [n][k])
  const int bk = t >> 3;
  const int bn = (t & 7) * 8;

  v8f acc = {};
  for (int k0 = 0; k0 < K; k0 += 32) {
    // stage A
    *(v4h*)&sa[am * LDA + ak] = *(const v4h*)&A[(size_t)arow * K + k0 + ak];
    // stage B transposed
    v4h w0 = *(const v4h*)&B[(size_t)(k0 + bk) * F + n0 + bn];
    v4h w1 = *(const v4h*)&B[(size_t)(k0 + bk) * F + n0 + bn + 4];
#pragma unroll
    for (int j = 0; j < 4; j++) {
      sb[(bn + j) * LDB + bk]     = w0[j];
      sb[(bn + 4 + j) * LDB + bk] = w1[j];
    }
    if (k0 + 32 < K) {
      __builtin_prefetch(&A[(size_t)arow * K + k0 + 32 + ak], 0, 1);
      __builtin_prefetch(&B[(size_t)(k0 + 32 + bk) * F + n0 + bn], 0, 1);
    }
    __syncthreads();

    // A fragment: row = mt*16+mn; halfs [8h .. 8h+7] -> elems 0..7, [16+8h ..] -> elems 8..15
    const _Float16* ap = &sa[(mt * 16 + mn) * LDA + 8 * half_sel];
    v8h alo = *(const v8h*)ap;
    v8h ahi = *(const v8h*)(ap + 16);
    // B fragment: column n = nt*16+mn, same K striping (B staged K-contiguous per column)
    const _Float16* bp = &sb[(nt * 16 + mn) * LDB + 8 * half_sel];
    v8h blo = *(const v8h*)bp;
    v8h bhi = *(const v8h*)(bp + 16);

    v16h a = __builtin_shufflevector(alo, ahi, 0,1,2,3,4,5,6,7,8,9,10,11,12,13,14,15);
    v16h b = __builtin_shufflevector(blo, bhi, 0,1,2,3,4,5,6,7,8,9,10,11,12,13,14,15);

    acc = __builtin_amdgcn_wmma_f32_16x16x32_f16(false, a, false, b, (short)0, acc,
                                                 false, false);
    __syncthreads();
  }

  // D layout: VGPR r -> row r + 8*half_sel, col = lane%16
  const int col   = n0 + nt * 16 + mn;
  const int rbase = m0 + mt * 16 + 8 * half_sel;
  const float bv  = bias ? bias[col] : 0.f;
#pragma unroll
  for (int r = 0; r < 8; r++) {
    int row = rbase + r;
    if (row < M) {
      float v = acc[r] + bv;
      if (relu) v = fmaxf(v, 0.f);
      C[(size_t)row * F + col] = v;
    }
  }
}

// ---------------- GCN message passing ----------------

// Hacc[i,f] = Hlin[i,f] * dis[i]^2   (self-loop contribution, also zero-inits accumulator)
__global__ void k_init_selfloop(const float* __restrict__ Hlin, const float* __restrict__ dis,
                                float* __restrict__ Hacc, int N) {
  int i = blockIdx.x * blockDim.x + threadIdx.x;
  int node = i >> 8;
  if (node < N) {
    float d = dis[node];
    Hacc[i] = Hlin[i] * d * d;
  }
}

// one block (=HID threads) per edge: Hacc[dst,f] += Hlin[src,f] * dis[src]*dis[dst]
__global__ __launch_bounds__(HID)
void k_edge_scatter(const int* __restrict__ src, const int* __restrict__ dst,
                    const float* __restrict__ dis,
                    const float* __restrict__ Hlin, float* __restrict__ Hacc, int E) {
  int e = blockIdx.x;
  int f = threadIdx.x;
  int s = src[e], d = dst[e];
  float w = dis[s] * dis[d];
  atomicAdd(&Hacc[(size_t)d * HID + f], Hlin[(size_t)s * HID + f] * w);
}

__global__ void k_bias_relu_f16(const float* __restrict__ Hacc, const float* __restrict__ b,
                                _Float16* __restrict__ out, int N) {
  int i = blockIdx.x * blockDim.x + threadIdx.x;
  if (i < N * HID) {
    float v = Hacc[i] + b[i & (HID - 1)];
    out[i] = (_Float16)fmaxf(v, 0.f);
  }
}

// ---------------- pooling + head ----------------

__global__ __launch_bounds__(HID)
void k_pool(const _Float16* __restrict__ h, const int* __restrict__ batch,
            float* __restrict__ gsum, float* __restrict__ cnt, int N) {
  int node = blockIdx.x;
  int f = threadIdx.x;
  int g = batch[node];
  atomicAdd(&gsum[(size_t)g * HID + f], (float)h[(size_t)node * HID + f]);
  if (f == 0) atomicAdd(&cnt[g], 1.0f);
}

__global__ void k_mean_f16(const float* __restrict__ gsum, const float* __restrict__ cnt,
                           _Float16* __restrict__ g16, int G) {
  int i = blockIdx.x * blockDim.x + threadIdx.x;
  if (i < G * HID) {
    float c = cnt[i >> 8];
    g16[i] = (_Float16)(gsum[i] / fmaxf(c, 1.f));
  }
}

// out[row] = gmid[row,:] . Wo + bo ; one wave32 per row, shuffle reduction
__global__ __launch_bounds__(256)
void k_gemv_out(const float* __restrict__ gmid, const float* __restrict__ Wo,
                const float* __restrict__ bo, float* __restrict__ out, int G) {
  int wid  = threadIdx.x >> 5;
  int lane = threadIdx.x & 31;
  int row  = blockIdx.x * 8 + wid;
  if (row >= G) return;
  float s = 0.f;
#pragma unroll
  for (int j = 0; j < 4; j++)
    s += gmid[(size_t)row * 128 + j * 32 + lane] * Wo[j * 32 + lane];
  for (int off = 16; off > 0; off >>= 1) s += __shfl_down(s, off, 32);
  if (lane == 0) out[row] = s + bo[0];
}

// ---------------- host launch ----------------

extern "C" void kernel_launch(void* const* d_in, const int* in_sizes, int n_in,
                              void* d_out, int out_size, void* d_ws, size_t ws_size,
                              hipStream_t stream) {
  const float* x   = (const float*)d_in[0];
  const int*   ei  = (const int*)d_in[1];
  const int*   bat = (const int*)d_in[2];
  const float* W1  = (const float*)d_in[3];
  const float* b1  = (const float*)d_in[4];
  const float* W2  = (const float*)d_in[5];
  const float* b2  = (const float*)d_in[6];
  const float* W3  = (const float*)d_in[7];
  const float* b3  = (const float*)d_in[8];
  const float* Wl  = (const float*)d_in[9];
  const float* bl  = (const float*)d_in[10];
  const float* Wo  = (const float*)d_in[11];
  const float* bo  = (const float*)d_in[12];

  const int N = in_sizes[0] / 64;   // 50000
  const int E = in_sizes[1] / 2;    // 800000
  const int G = out_size;           // 2048
  const int H = HID, HL = 128, F0 = 64;
  const int* src = ei;
  const int* dst = ei + E;

  char* ws = (char*)d_ws;
  size_t off = 0;
  auto alloc = [&](size_t bytes) -> char* {
    char* p = ws + off;
    off += (bytes + 255) & ~(size_t)255;
    return p;
  };

  float*    deg  = (float*)alloc((size_t)N * 4);
  float*    dis  = (float*)alloc((size_t)N * 4);
  _Float16* W1h  = (_Float16*)alloc((size_t)F0 * H * 2);
  _Float16* W2h  = (_Float16*)alloc((size_t)H * H * 2);
  _Float16* W3h  = (_Float16*)alloc((size_t)H * H * 2);
  _Float16* Wlh  = (_Float16*)alloc((size_t)H * HL * 2);
  _Float16* Ha   = (_Float16*)alloc((size_t)N * H * 2);   // also holds x(f16) for layer 1
  _Float16* Hb   = (_Float16*)alloc((size_t)N * H * 2);
  float*    Hlin = (float*)alloc((size_t)N * H * 4);
  float*    Hacc = (float*)alloc((size_t)N * H * 4);
  float*    gsum = (float*)alloc((size_t)G * H * 4);
  float*    cnt  = (float*)alloc((size_t)G * 4);
  _Float16* g16  = (_Float16*)alloc((size_t)G * H * 2);
  float*    gmid = (float*)alloc((size_t)G * HL * 4);
  (void)ws_size; (void)n_in;

  const int TB = 256;
  const int mb = (N + 31) / 32;     // GEMM M-blocks for node GEMMs

  // degree / normalization
  k_set_f32<<<(N + TB - 1) / TB, TB, 0, stream>>>(deg, 1.0f, N);           // self-loop
  k_deg_edges<<<(E + TB - 1) / TB, TB, 0, stream>>>(dst, deg, E);
  k_rsqrt<<<(N + TB - 1) / TB, TB, 0, stream>>>(deg, dis, N);

  // f16 conversions
  k_f32_to_f16<<<(F0 * H + TB - 1) / TB, TB, 0, stream>>>(W1, W1h, F0 * H);
  k_f32_to_f16<<<(H * H + TB - 1) / TB, TB, 0, stream>>>(W2, W2h, H * H);
  k_f32_to_f16<<<(H * H + TB - 1) / TB, TB, 0, stream>>>(W3, W3h, H * H);
  k_f32_to_f16<<<(H * HL + TB - 1) / TB, TB, 0, stream>>>(Wl, Wlh, H * HL);
  k_f32_to_f16<<<(N * F0 + TB - 1) / TB, TB, 0, stream>>>(x, Ha, N * F0);

  // ---- layer 1: A = x(f16)[N x 64] ----
  k_gemm_wmma<<<dim3(mb, H / 64), TB, 0, stream>>>(Ha, W1h, nullptr, Hlin, N, F0, H, 0);
  k_init_selfloop<<<((size_t)N * H + TB - 1) / TB, TB, 0, stream>>>(Hlin, dis, Hacc, N);
  k_edge_scatter<<<E, H, 0, stream>>>(src, dst, dis, Hlin, Hacc, E);
  k_bias_relu_f16<<<((size_t)N * H + TB - 1) / TB, TB, 0, stream>>>(Hacc, b1, Hb, N);

  // ---- layer 2 ----
  k_gemm_wmma<<<dim3(mb, H / 64), TB, 0, stream>>>(Hb, W2h, nullptr, Hlin, N, H, H, 0);
  k_init_selfloop<<<((size_t)N * H + TB - 1) / TB, TB, 0, stream>>>(Hlin, dis, Hacc, N);
  k_edge_scatter<<<E, H, 0, stream>>>(src, dst, dis, Hlin, Hacc, E);
  k_bias_relu_f16<<<((size_t)N * H + TB - 1) / TB, TB, 0, stream>>>(Hacc, b2, Ha, N);

  // ---- layer 3 ----
  k_gemm_wmma<<<dim3(mb, H / 64), TB, 0, stream>>>(Ha, W3h, nullptr, Hlin, N, H, H, 0);
  k_init_selfloop<<<((size_t)N * H + TB - 1) / TB, TB, 0, stream>>>(Hlin, dis, Hacc, N);
  k_edge_scatter<<<E, H, 0, stream>>>(src, dst, dis, Hlin, Hacc, E);
  k_bias_relu_f16<<<((size_t)N * H + TB - 1) / TB, TB, 0, stream>>>(Hacc, b3, Hb, N);

  // ---- global mean pool ----
  k_set_f32<<<(G * H + TB - 1) / TB, TB, 0, stream>>>(gsum, 0.f, G * H);
  k_set_f32<<<(G + TB - 1) / TB, TB, 0, stream>>>(cnt, 0.f, G);
  k_pool<<<N, H, 0, stream>>>(Hb, bat, gsum, cnt, N);
  k_mean_f16<<<(G * H + TB - 1) / TB, TB, 0, stream>>>(gsum, cnt, g16, G);

  // ---- head: relu(g @ Wl + bl) @ Wo + bo ----
  k_gemm_wmma<<<dim3(G / 32, HL / 64), TB, 0, stream>>>(g16, Wlh, bl, gmid, G, H, HL, 1);
  k_gemv_out<<<(G + 7) / 8, TB, 0, stream>>>(gmid, Wo, bo, (float*)d_out, G);
}
